// Glimpse_81389630259572
// MI455X (gfx1250) — matrix operand order repaired and checked by reading, not compiled, not run on hardware
//
#include <hip/hip_runtime.h>

#define BQ 128
#define SQ 4096
#define DQ 256
#define HQ 8
#define SPLITQ 4
#define SEGQ (SQ / SPLITQ)   // 1024
#define TSQ 64
#define NTILES (SEGQ / TSQ)  // 16
#define SCALEQ 0.17677669529663687f  // 1/sqrt(32)

typedef _Float16 half16 __attribute__((ext_vector_type(16)));
typedef _Float16 half2v __attribute__((ext_vector_type(2)));
typedef float v8f __attribute__((ext_vector_type(8)));
typedef float f4v __attribute__((ext_vector_type(4)));

// ---- LDS byte offsets (all 16B aligned) ----
#define TROW 528                   // (256+8) f16, 132 dw -> stride 4 banks: conflict-free frags
#define OFF_T 0
#define SZ_T (64 * TROW)           // 33792
#define TTROW 144                  // (64+8) f16, 36 dw
#define OFF_TT (OFF_T + SZ_T)
#define SZ_TT (256 * TTROW)        // 36864
#define PROW 144
#define OFF_P (OFF_TT + SZ_TT)
#define SZ_P (16 * PROW)
#define WROW 528
#define OFF_W (OFF_P + SZ_P)
#define SZ_W (16 * WROW)
#define OFF_QV (OFF_W + SZ_W)      // 256 f32
#define OFF_C  (OFF_QV + 1024)     // 16 f32
#define OFF_M  (OFF_C + 64)
#define OFF_L  (OFF_M + 64)
#define OFF_SC (OFF_L + 64)
#define OFF_RMAX (OFF_SC + 64)     // 8*16 f32
#define OFF_RSUM (OFF_RMAX + 512)
#define SMEM_TOTAL (OFF_RSUM + 512)  // 83712 bytes

// A fragment (16x32 f16, M rows in lanes): elems 0..7 = K+{0..7}, 8..15 = K+{16..23}
__device__ __forceinline__ half16 ldsLoadA(const char* p) {
  union { half16 h; uint4 q[2]; } f;
  f.q[0] = *(const uint4*)p;
  f.q[1] = *(const uint4*)(p + 32);
  return f.h;
}
// A fragment with independent chunk offsets (for XOR-swizzled Tt)
__device__ __forceinline__ half16 ldsLoadA2(const char* rowp, int off0, int off1) {
  union { half16 h; uint4 q[2]; } f;
  f.q[0] = *(const uint4*)(rowp + off0);
  f.q[1] = *(const uint4*)(rowp + off1);
  return f.h;
}
// B fragment (32x16 f16, N col per lane): elems 0..7 = K+{0..7}, 8..15 = K+{16..23}
__device__ __forceinline__ half16 ldsLoadB(const char* p) {
  union { half16 h; uint4 q[2]; } f;
  f.q[0] = *(const uint4*)p;
  f.q[1] = *(const uint4*)(p + 32);
  return f.h;
}

__global__ void __launch_bounds__(256) glimpse_pass1(
    const float* __restrict__ query, const float* __restrict__ target,
    const unsigned char* __restrict__ mask,
    const float* __restrict__ Wq, const float* __restrict__ bq,
    const float* __restrict__ Wk, const float* __restrict__ bk,
    float* __restrict__ alphaOut,
    float* __restrict__ wsM, float* __restrict__ wsL, float* __restrict__ wsU)
{
  extern __shared__ char smem[];
  const int tid  = threadIdx.x;
  const int lane = tid & 31;
  const int w    = tid >> 5;
  const int b    = blockIdx.x / SPLITQ;
  const int sp   = blockIdx.x % SPLITQ;
  const int sBase0 = sp * SEGQ;
  const float NEGINF = -__builtin_inff();

  // ---- q[b] projection: thread t computes q_t ----
  {
    const float* qrow = query + (size_t)b * DQ;
    const float* wrow = Wq + (size_t)tid * DQ;
    float acc = bq[tid];
    #pragma unroll 8
    for (int j = 0; j < DQ / 4; ++j) {
      f4v qv4 = ((const f4v*)qrow)[j];
      f4v wv4 = ((const f4v*)wrow)[j];
      acc += qv4.x * wv4.x + qv4.y * wv4.y + qv4.z * wv4.z + qv4.w * wv4.w;
    }
    ((float*)(smem + OFF_QV))[tid] = acc;
  }
  __syncthreads();
  // ---- w_eff[h][d] = SCALE * sum_hd q[hd]*Wk[hd][d]; c[h]; init m,l ----
  {
    const float* qv = (const float*)(smem + OFF_QV);
    float acc[8];
    #pragma unroll
    for (int h = 0; h < 8; ++h) acc[h] = 0.f;
    const int d = tid;
    #pragma unroll 8
    for (int hd = 0; hd < 256; ++hd)
      acc[hd >> 5] += qv[hd] * Wk[(size_t)hd * DQ + d];
    #pragma unroll
    for (int h = 0; h < 16; ++h) {
      _Float16 v = (h < 8) ? (_Float16)(SCALEQ * acc[h]) : (_Float16)0.f;
      *(_Float16*)(smem + OFF_W + h * WROW + d * 2) = v;
    }
    if (tid < 16) {
      float c = 0.f;
      if (tid < 8) {
        #pragma unroll
        for (int j = 0; j < 32; ++j) c += qv[tid * 32 + j] * bk[tid * 32 + j];
        c *= SCALEQ;
      }
      ((float*)(smem + OFF_C))[tid] = c;
      ((float*)(smem + OFF_M))[tid] = NEGINF;
      ((float*)(smem + OFF_L))[tid] = 0.f;
    }
  }
  __syncthreads();

  v8f u0 = {};  // u^T accumulator, rows D = w*32 + ..., cols = heads
  v8f u1 = {};  // rows D = w*32+16 + ...

  for (int t = 0; t < NTILES; ++t) {
    const int sAbs = sBase0 + t * TSQ;
    // ---- Phase A: global fp32 -> f16 LDS (row-major + XOR-swizzled transposed) ----
    {
      const int g  = tid >> 3;            // 0..31 -> row pair
      const int c0 = (tid & 7) * 32;      // col chunk
      const int r0 = 2 * g, r1 = r0 + 1;
      const float* gp0 = target + ((size_t)b * SQ + sAbs + r0) * DQ + c0;
      const float* gp1 = gp0 + DQ;
      _Float16 h0[32], h1[32];
      #pragma unroll
      for (int j = 0; j < 8; ++j) {
        f4v v = __builtin_nontemporal_load((const f4v*)gp0 + j);
        h0[4*j+0] = (_Float16)v.x; h0[4*j+1] = (_Float16)v.y;
        h0[4*j+2] = (_Float16)v.z; h0[4*j+3] = (_Float16)v.w;
      }
      #pragma unroll
      for (int j = 0; j < 8; ++j) {
        f4v v = __builtin_nontemporal_load((const f4v*)gp1 + j);
        h1[4*j+0] = (_Float16)v.x; h1[4*j+1] = (_Float16)v.y;
        h1[4*j+2] = (_Float16)v.z; h1[4*j+3] = (_Float16)v.w;
      }
      #pragma unroll
      for (int gg = 0; gg < 4; ++gg) {
        union { _Float16 h[8]; uint4 q; } u;
        #pragma unroll
        for (int e = 0; e < 8; ++e) u.h[e] = h0[8*gg + e];
        *(uint4*)(smem + OFF_T + r0 * TROW + c0 * 2 + gg * 16) = u.q;
      }
      #pragma unroll
      for (int gg = 0; gg < 4; ++gg) {
        union { _Float16 h[8]; uint4 q; } u;
        #pragma unroll
        for (int e = 0; e < 8; ++e) u.h[e] = h1[8*gg + e];
        *(uint4*)(smem + OFF_T + r1 * TROW + c0 * 2 + gg * 16) = u.q;
      }
      // transposed store: s-chunk XOR swizzle ((s>>3) ^ ((col>>5)&7)) keeps data
      // indices constant (no register-array selects) and spreads banks.
      const int cXor  = tid & 7;                                   // (col>>5)&7
      const int physS = (((r0 >> 3) ^ cXor) << 3) | (r0 & 7);      // swizzled s
      char* tbase = smem + OFF_TT + c0 * TTROW + physS * 2;
      #pragma unroll
      for (int j = 0; j < 32; ++j) {
        union { half2v h; unsigned u32; } p;
        p.h[0] = h0[j]; p.h[1] = h1[j];
        *(unsigned*)(tbase + j * TTROW) = p.u32;
      }
    }
    __syncthreads();

    // ---- Phase B: scores = T @ w_eff^T  (waves 0..3, 16 rows each) ----
    float scr[8];
    if (w < 4) {
      v8f c = {};
      const int Mb = w * 16;
      const int aoff = (lane & 16) ? 16 : 0;   // bytes: ksub*2
      const int boff = (lane >> 4) << 4;       // bytes: 8 elems * 2
      #pragma unroll
      for (int k = 0; k < 8; ++k) {
        half16 af = ldsLoadA(smem + OFF_T + (Mb + (lane & 15)) * TROW + k * 64 + aoff);
        half16 bf = ldsLoadB(smem + OFF_W + (lane & 15) * WROW + k * 64 + boff);
        c = __builtin_amdgcn_wmma_f32_16x16x32_f16(false, af, false, bf, (short)0, c, false, false);
      }
      const int head  = lane & 15;
      const float cb  = ((const float*)(smem + OFF_C))[head];
      const int srow  = Mb + ((lane >> 4) << 3);
      float mx = NEGINF;
      #pragma unroll
      for (int i = 0; i < 8; ++i) {
        float s = c[i] + cb;
        if (mask[(size_t)b * SQ + sAbs + srow + i]) s = NEGINF;
        scr[i] = s;
        mx = fmaxf(mx, s);
        if (head < 8)
          alphaOut[((size_t)b * HQ + head) * SQ + sAbs + srow + i] = s;
      }
      ((float*)(smem + OFF_RMAX))[(w * 2 + (lane >> 4)) * 16 + head] = mx;
    }
    __syncthreads();
    if (tid < 16) {  // m update + rescale factor
      float* mbuf = (float*)(smem + OFF_M);
      float* scb  = (float*)(smem + OFF_SC);
      const float* rm = (const float*)(smem + OFF_RMAX);
      float mo = mbuf[tid], mn = mo;
      #pragma unroll
      for (int j = 0; j < 8; ++j) mn = fmaxf(mn, rm[j * 16 + tid]);
      scb[tid] = (mo < -3.0e37f) ? 0.f : __expf(mo - mn);
      mbuf[tid] = mn;
    }
    __syncthreads();
    // rescale u accumulators (all waves; column = head = lane&15)
    {
      const float s = ((const float*)(smem + OFF_SC))[lane & 15];
      #pragma unroll
      for (int i = 0; i < 8; ++i) { u0[i] *= s; u1[i] *= s; }
    }
    // ---- p = exp(score - m) -> P_lds[h][s], partial sums ----
    if (w < 4) {
      const int head = lane & 15;
      const int srow = (w * 16) + ((lane >> 4) << 3);
      const float mn = ((const float*)(smem + OFF_M))[head];
      float ps = 0.f;
      #pragma unroll
      for (int i = 0; i < 8; i += 2) {
        float p0 = (scr[i]     < -3.0e37f) ? 0.f : __expf(scr[i]     - mn);
        float p1 = (scr[i + 1] < -3.0e37f) ? 0.f : __expf(scr[i + 1] - mn);
        ps += p0 + p1;
        union { half2v h; unsigned u32; } pk;
        pk.h[0] = (_Float16)p0; pk.h[1] = (_Float16)p1;
        *(unsigned*)(smem + OFF_P + head * PROW + (srow + i) * 2) = pk.u32;
      }
      ((float*)(smem + OFF_RSUM))[(w * 2 + (lane >> 4)) * 16 + head] = ps;
    }
    __syncthreads();
    if (tid < 16) {  // l = l*scale + sum(p)
      float* lbuf = (float*)(smem + OFF_L);
      const float* rs = (const float*)(smem + OFF_RSUM);
      float lt = 0.f;
      #pragma unroll
      for (int j = 0; j < 8; ++j) lt += rs[j * 16 + tid];
      lbuf[tid] = lbuf[tid] * ((const float*)(smem + OFF_SC))[tid] + lt;
    }
    // ---- Phase C: u^T[D,h] += Tt @ P^T  (all waves, 2 M-tiles) ----
    // Tt s-chunks are XOR-swizzled by (Drow>>5)&7 == w (constant per wave).
    {
      const int sub  = (lane >> 4) & 1;        // logical chunk offset 0/1
      const int boff = (lane >> 4) << 4;
      const char* rp0 = smem + OFF_TT + (w * 32 + (lane & 15)) * TTROW;
      const char* rp1 = rp0 + 16 * TTROW;
      #pragma unroll
      for (int k = 0; k < 2; ++k) {
        const int ch0 = k * 4 + sub;           // chunk for elems 0..7
        const int ch1 = ch0 + 2;               // chunk for elems 8..15
        const int off0 = ((ch0 ^ w) << 4);
        const int off1 = ((ch1 ^ w) << 4);
        half16 bf = ldsLoadB(smem + OFF_P + (lane & 15) * PROW + k * 64 + boff);
        half16 a0 = ldsLoadA2(rp0, off0, off1);
        u0 = __builtin_amdgcn_wmma_f32_16x16x32_f16(false, a0, false, bf, (short)0, u0, false, false);
        half16 a1 = ldsLoadA2(rp1, off0, off1);
        u1 = __builtin_amdgcn_wmma_f32_16x16x32_f16(false, a1, false, bf, (short)0, u1, false, false);
      }
    }
    __syncthreads();
  }

  // ---- write split results ----
  if (tid < 8) {
    wsM[(b * SPLITQ + sp) * 8 + tid] = ((const float*)(smem + OFF_M))[tid];
    wsL[(b * SPLITQ + sp) * 8 + tid] = ((const float*)(smem + OFF_L))[tid];
  }
  {
    const int head = lane & 15;
    if (head < 8) {
      float* up = wsU + (((size_t)(b * SPLITQ + sp)) * 8 + head) * 256;
      const int dB = w * 32 + ((lane >> 4) << 3);
      #pragma unroll
      for (int i = 0; i < 8; ++i) up[dB + i] = u0[i];
      #pragma unroll
      for (int i = 0; i < 8; ++i) up[dB + 16 + i] = u1[i];
    }
  }
}

// ---- combine splits + output projection (one block per b) ----
__global__ void __launch_bounds__(256) glimpse_pass2(
    const float* __restrict__ wsM, const float* __restrict__ wsL,
    const float* __restrict__ wsU,
    float* __restrict__ wsMg, float* __restrict__ wsLg,
    const float* __restrict__ Wv, const float* __restrict__ bv,
    const float* __restrict__ Wo, const float* __restrict__ bo,
    float* __restrict__ outPtr)
{
  __shared__ float lg[8], fac[SPLITQ * 8], teff[8][260], att[256];
  const int b = blockIdx.x, tid = threadIdx.x;
  if (tid < 8) {
    float m = -__builtin_inff();
    for (int sp = 0; sp < SPLITQ; ++sp) m = fmaxf(m, wsM[(b * SPLITQ + sp) * 8 + tid]);
    float l = 0.f;
    for (int sp = 0; sp < SPLITQ; ++sp) {
      float mm = wsM[(b * SPLITQ + sp) * 8 + tid];
      float f = (mm < -3.0e37f) ? 0.f : __expf(mm - m);
      fac[sp * 8 + tid] = f;
      l += wsL[(b * SPLITQ + sp) * 8 + tid] * f;
    }
    lg[tid] = l; wsMg[b * 8 + tid] = m; wsLg[b * 8 + tid] = l;
  }
  __syncthreads();
  {
    const int d = tid;
    #pragma unroll
    for (int h = 0; h < 8; ++h) {
      float acc = 0.f;
      for (int sp = 0; sp < SPLITQ; ++sp)
        acc += wsU[(((size_t)(b * SPLITQ + sp)) * 8 + h) * 256 + d] * fac[sp * 8 + h];
      teff[h][d] = acc / lg[h];
    }
  }
  __syncthreads();
  {
    const int hd = tid, head = tid >> 5;
    float acc = bv[hd];
    const float* wr = Wv + (size_t)hd * 256;
    #pragma unroll 8
    for (int j = 0; j < 256; ++j) acc += teff[head][j] * wr[j];
    att[hd] = acc;
  }
  __syncthreads();
  {
    float o = bo[tid];
    const float* wr = Wo + (size_t)tid * 256;
    #pragma unroll 8
    for (int j = 0; j < 256; ++j) o += att[j] * wr[j];
    outPtr[(size_t)b * 256 + tid] = o;
  }
}

// ---- normalize alpha in place: exp(raw - m)/l ----
__global__ void __launch_bounds__(256) glimpse_pass3(
    float* __restrict__ alpha, const float* __restrict__ wsMg,
    const float* __restrict__ wsLg)
{
  const size_t i4 = (size_t)blockIdx.x * 256 + threadIdx.x;
  const size_t base = i4 * 4;
  const int b = (int)(base >> 15);       // / (H*S)
  const int h = (int)(base >> 12) & 7;   // / S % H
  const float m = wsMg[b * 8 + h];
  const float li = 1.0f / wsLg[b * 8 + h];
  f4v r = ((f4v*)alpha)[i4];
  r.x = (r.x < -3.0e37f) ? 0.f : __expf(r.x - m) * li;
  r.y = (r.y < -3.0e37f) ? 0.f : __expf(r.y - m) * li;
  r.z = (r.z < -3.0e37f) ? 0.f : __expf(r.z - m) * li;
  r.w = (r.w < -3.0e37f) ? 0.f : __expf(r.w - m) * li;
  ((f4v*)alpha)[i4] = r;
}

extern "C" void kernel_launch(void* const* d_in, const int* in_sizes, int n_in,
                              void* d_out, int out_size, void* d_ws, size_t ws_size,
                              hipStream_t stream) {
  (void)in_sizes; (void)n_in; (void)out_size; (void)ws_size;
  const float* query = (const float*)d_in[0];
  const float* target = (const float*)d_in[1];
  const unsigned char* mask = (const unsigned char*)d_in[2];
  const float* Wq = (const float*)d_in[3];
  const float* bq = (const float*)d_in[4];
  const float* Wk = (const float*)d_in[5];
  const float* bk = (const float*)d_in[6];
  const float* Wv = (const float*)d_in[7];
  const float* bv = (const float*)d_in[8];
  const float* Wo = (const float*)d_in[9];
  const float* bo = (const float*)d_in[10];

  float* alpha  = (float*)d_out;
  float* outPtr = alpha + (size_t)BQ * HQ * SQ;

  float* wsM  = (float*)d_ws;                 // B*SPLIT*8
  float* wsL  = wsM + BQ * SPLITQ * 8;
  float* wsMg = wsL + BQ * SPLITQ * 8;        // B*8
  float* wsLg = wsMg + BQ * 8;
  float* wsU  = wsLg + BQ * 8;                // B*SPLIT*8*256

  (void)hipFuncSetAttribute((const void*)glimpse_pass1,
                            hipFuncAttributeMaxDynamicSharedMemorySize, SMEM_TOTAL);
  glimpse_pass1<<<dim3(BQ * SPLITQ), dim3(256), SMEM_TOTAL, stream>>>(
      query, target, mask, Wq, bq, Wk, bk, alpha, wsM, wsL, wsU);
  glimpse_pass2<<<dim3(BQ), dim3(256), 0, stream>>>(
      wsM, wsL, wsU, wsMg, wsLg, Wv, bv, Wo, bo, outPtr);
  glimpse_pass3<<<dim3((BQ * HQ * SQ) / 1024), dim3(256), 0, stream>>>(
      alpha, wsMg, wsLg);
}
